// Reverb_37211596653118
// MI455X (gfx1250) — matrix operand order repaired and checked by reading, not compiled, not run on hardware
//
#include <hip/hip_runtime.h>
#include <hip/hip_bf16.h>
#include <stdint.h>

#define N_FFTC 384
#define HOPC   192
#define FREQC  193
#define STEPSC 311
#define FRC    1366          // frames per (b, stereo)
#define FRPC   1376          // padded to multiple of 16
#define BATCH  8
#define TLEN   262144
#define CHC    386           // 2*FREQ
#define NPAD_STFT 400        // 386 -> 25*16
#define KPAD_ISTFT 416       // 386 -> 13*32
#define MROWS (16 * FRPC)    // 22016 = B*2*FRP
#define MTILES (MROWS / 16)  // 1376
#define OUTLEN 262080        // (FR-1)*HOP

typedef __attribute__((ext_vector_type(16))) _Float16 v16h;
typedef __attribute__((ext_vector_type(8)))  _Float16 v8h;
typedef __attribute__((ext_vector_type(8)))  float    v8f;
typedef _Float16 f16_t;

#define TWO_PI 6.28318530717958647692f

__device__ __forceinline__ float softplusf(float x) {
  return (x > 20.0f) ? x : log1pf(expf(x));
}

__device__ __forceinline__ v16h cat8(v8h a0, v8h a1) {
  return __builtin_shufflevector(a0, a1, 0,1,2,3,4,5,6,7,8,9,10,11,12,13,14,15);
}

// ---------------------------------------------------------------------------
// Stage 0: IR weights  wr/wi[ch=386][311], stereo-mixed, stored UNFLIPPED
// (conv stage does y[t] = sum_d H[t-d] * ir[d], equivalent to the reference's
// kernel-flip + valid cross-correlation with left zero pad)
// ---------------------------------------------------------------------------
__global__ void k_weights(const float* __restrict__ lmr,
                          const float* __restrict__ lmdr,
                          const float* __restrict__ ang,
                          float* __restrict__ wr, float* __restrict__ wi) {
  int idx = blockIdx.x * blockDim.x + threadIdx.x;
  if (idx >= FREQC * STEPSC) return;
  int q = idx / STEPSC, t = idx % STEPSC;
  float lm0 = -softplusf(lmr[q]),         lm1 = -softplusf(lmr[FREQC + q]);
  float ld0 = -softplusf(lmdr[q]),        ld1 = -softplusf(lmdr[FREQC + q]);
  float m0 = expf(lm0 + ld0 * (float)t),  m1 = expf(lm1 + ld1 * (float)t);
  float a0 = ang[(size_t)q * STEPSC + t];
  float a1 = ang[(size_t)(FREQC + q) * STEPSC + t];
  float r0 = m0 * cosf(a0), i0 = m0 * sinf(a0);
  float r1 = m1 * cosf(a1), i1 = m1 * sinf(a1);
  const float is2 = 0.70710678118654752f;
  wr[(size_t)q * STEPSC + t]            = (r0 + r1) * is2;
  wr[(size_t)(FREQC + q) * STEPSC + t]  = (r0 - r1) * is2;
  wi[(size_t)q * STEPSC + t]            = (i0 + i1) * is2;
  wi[(size_t)(FREQC + q) * STEPSC + t]  = (i0 - i1) * is2;
}

// ---------------------------------------------------------------------------
// Stage 1: analysis DFT basis (Hann window folded in), stored [j][ns] so each
// lane's 16 K-elements form two runs of 8 (k=8h.., 16+8h..).  j<193: win*cos ; 193..385: -win*sin
// ---------------------------------------------------------------------------
__global__ void k_stft_basis(f16_t* __restrict__ bst) {
  int idx = blockIdx.x * blockDim.x + threadIdx.x;
  if (idx >= NPAD_STFT * N_FFTC) return;
  int j = idx / N_FFTC, ns = idx % N_FFTC;
  float v = 0.0f;
  if (j < CHC) {
    int q = (j < FREQC) ? j : j - FREQC;
    int r = (q * ns) % N_FFTC;                     // exact angle reduction
    float angv = TWO_PI * (float)r / (float)N_FFTC;
    float win = 0.5f * (1.0f - cosf(TWO_PI * (float)ns / (float)N_FFTC));
    v = (j < FREQC) ? win * cosf(angv) : -win * sinf(angv);
  }
  bst[idx] = (f16_t)v;
}

// Synthesis basis: frames[n] = sum_k c_k/N * (Yr cos - Yi sin) * win[n]
__global__ void k_istft_basis(f16_t* __restrict__ ibt) {
  int idx = blockIdx.x * blockDim.x + threadIdx.x;
  if (idx >= N_FFTC * KPAD_ISTFT) return;
  int ns = idx / KPAD_ISTFT, k = idx % KPAD_ISTFT;
  float v = 0.0f;
  if (k < CHC) {
    int q = (k < FREQC) ? k : k - FREQC;
    float c = (q == 0 || q == N_FFTC / 2) ? 1.0f : 2.0f;
    int r = (q * ns) % N_FFTC;
    float angv = TWO_PI * (float)r / (float)N_FFTC;
    float win = 0.5f * (1.0f - cosf(TWO_PI * (float)ns / (float)N_FFTC));
    float s = win * c * (1.0f / (float)N_FFTC);
    v = (k < FREQC) ? s * cosf(angv) : -s * sinf(angv);
  }
  ibt[idx] = (f16_t)v;
}

// ---------------------------------------------------------------------------
// Stage 2: frame extraction with reflect padding -> f16 A-matrix [MROWS][384]
// ---------------------------------------------------------------------------
__global__ void k_frames(const float* __restrict__ x, f16_t* __restrict__ fa) {
  int idx = blockIdx.x * blockDim.x + threadIdx.x;
  if (idx >= MROWS * N_FFTC) return;
  int m = idx / N_FFTC, ns = idx % N_FFTC;
  int f = m % FRPC, bs = m / FRPC;
  float v = 0.0f;
  if (f < FRC) {
    int j = f * HOPC + ns - HOPC;
    if (j < 0) j = -j;
    else if (j >= TLEN) j = 2 * TLEN - 2 - j;
    v = x[(size_t)bs * TLEN + j];
  }
  fa[idx] = (f16_t)v;
}

// ---------------------------------------------------------------------------
// Stage 3: STFT as WMMA GEMM.
// Block: 256 threads = 8 waves; grid.y = N-tile; each wave owns 4 M-tiles.
// B column (16 cols x 384 K, 12KB f16) staged once per block into LDS;
// each wave runs 4 independent WMMA chains off one LDS B fragment per k-step.
// ---------------------------------------------------------------------------
__global__ __launch_bounds__(256)
void k_stft_wmma(const f16_t* __restrict__ fa, const f16_t* __restrict__ bst,
                 float* __restrict__ Hr, float* __restrict__ Hi) {
  __shared__ f16_t bsh[16 * N_FFTC];                 // 12 KB
  const int nt = blockIdx.y;
  // cooperative B stage: 16*384 halves = 768 uint4
  {
    const uint4* src = (const uint4*)(bst + (size_t)nt * 16 * N_FFTC);
    uint4* dst = (uint4*)bsh;
    for (int i = threadIdx.x; i < (16 * N_FFTC) / 8; i += 256) dst[i] = src[i];
  }
  int wave = threadIdx.x >> 5;
  int lane = threadIdx.x & 31;
  int hid = lane >> 4, lr = lane & 15;
  int mt0 = (blockIdx.x * 8 + wave) * 4;             // 4 consecutive M-tiles
  const f16_t* arow = fa + (size_t)(mt0 * 16 + lr) * N_FFTC + hid * 8;
  // warm L2/L0 with the A rows while B staging completes
  __builtin_prefetch(arow, 0, 3);
  __builtin_prefetch(arow + 16 * N_FFTC, 0, 3);
  __builtin_prefetch(arow + 32 * N_FFTC, 0, 3);
  __builtin_prefetch(arow + 48 * N_FFTC, 0, 3);
  __syncthreads();

  const f16_t* bsrc = bsh + (size_t)lr * N_FFTC + hid * 8;
  v8f acc0 = {}, acc1 = {}, acc2 = {}, acc3 = {};
#pragma unroll
  for (int kt = 0; kt < N_FFTC / 32; ++kt) {
    v16h b = cat8(*(const v8h*)(bsrc + kt * 32), *(const v8h*)(bsrc + kt * 32 + 16));
    const f16_t* ak = arow + kt * 32;
    v16h a0 = cat8(*(const v8h*)(ak),                  *(const v8h*)(ak + 16));
    v16h a1 = cat8(*(const v8h*)(ak + 16 * N_FFTC),    *(const v8h*)(ak + 16 * N_FFTC + 16));
    v16h a2 = cat8(*(const v8h*)(ak + 32 * N_FFTC),    *(const v8h*)(ak + 32 * N_FFTC + 16));
    v16h a3 = cat8(*(const v8h*)(ak + 48 * N_FFTC),    *(const v8h*)(ak + 48 * N_FFTC + 16));
    acc0 = __builtin_amdgcn_wmma_f32_16x16x32_f16(false, a0, false, b, (short)0, acc0, false, false);
    acc1 = __builtin_amdgcn_wmma_f32_16x16x32_f16(false, a1, false, b, (short)0, acc1, false, false);
    acc2 = __builtin_amdgcn_wmma_f32_16x16x32_f16(false, a2, false, b, (short)0, acc2, false, false);
    acc3 = __builtin_amdgcn_wmma_f32_16x16x32_f16(false, a3, false, b, (short)0, acc3, false, false);
  }
  int j = nt * 16 + lr;                              // output column
  if (j >= CHC) return;                              // padded N columns
  int q = (j < FREQC) ? j : j - FREQC;
  float* hbase = (j < FREQC) ? Hr : Hi;
  int row0 = hid * 8;
  v8f accs[4] = {acc0, acc1, acc2, acc3};
#pragma unroll
  for (int u = 0; u < 4; ++u) {
    int m0 = (mt0 + u) * 16;
    int f0 = m0 % FRPC;                              // 16-aligned within one (b,s)
    int bs = m0 / FRPC;
    int bb = bs >> 1, ss = bs & 1;
    float* dst = hbase + ((size_t)(ss * FREQC + q) * BATCH + bb) * FRPC;
#pragma unroll
    for (int v = 0; v < 8; ++v) {
      int f = f0 + row0 + v;
      if (f < FRC) dst[f] = accs[u][v];
    }
  }
}

// ---------------------------------------------------------------------------
// Stage 4: depthwise complex conv (311 taps, causal), fully LDS-resident:
// taps (2.5KB) + H time series (11KB) staged per block, inner loop is ds+FMA.
// Writes straight into the f16 ISTFT A-matrix Ycat [MROWS][416].
// ---------------------------------------------------------------------------
__global__ __launch_bounds__(256)
void k_conv(const float* __restrict__ Hr, const float* __restrict__ Hi,
            const float* __restrict__ wr, const float* __restrict__ wi,
            f16_t* __restrict__ ycat) {
  __shared__ float swr[STEPSC];
  __shared__ float swi[STEPSC];
  __shared__ float shr[FRC];
  __shared__ float shi[FRC];
  int ch = blockIdx.x / BATCH;
  int b  = blockIdx.x % BATCH;
  const float* hr = Hr + ((size_t)ch * BATCH + b) * FRPC;
  const float* hi = Hi + ((size_t)ch * BATCH + b) * FRPC;
  for (int i = threadIdx.x; i < STEPSC; i += blockDim.x) {
    swr[i] = wr[(size_t)ch * STEPSC + i];
    swi[i] = wi[(size_t)ch * STEPSC + i];
  }
  for (int i = threadIdx.x; i < FRC; i += blockDim.x) {
    shr[i] = hr[i];
    shi[i] = hi[i];
  }
  __syncthreads();
  int ss = ch / FREQC, q = ch % FREQC;
  for (int t = threadIdx.x; t < FRC; t += blockDim.x) {
    float yr = 0.0f, yi = 0.0f;
    int dmax = (t < STEPSC - 1) ? t : (STEPSC - 1);
    for (int d = 0; d <= dmax; ++d) {
      float ar = shr[t - d], ai = shi[t - d];
      float cr = swr[d],     ci = swi[d];
      yr = fmaf(ar, cr, fmaf(-ai, ci, yr));
      yi = fmaf(ar, ci, fmaf( ai, cr, yi));
    }
    size_t m = ((size_t)(b * 2 + ss)) * FRPC + t;
    ycat[m * KPAD_ISTFT + q]         = (f16_t)yr;
    ycat[m * KPAD_ISTFT + FREQC + q] = (f16_t)yi;
  }
}

// ---------------------------------------------------------------------------
// Stage 5: ISTFT as WMMA GEMM (same scheme; K=416, N=384, D -> f16 frames).
// ---------------------------------------------------------------------------
__global__ __launch_bounds__(256)
void k_istft_wmma(const f16_t* __restrict__ ycat, const f16_t* __restrict__ ibt,
                  f16_t* __restrict__ fout) {
  __shared__ f16_t bsh[16 * KPAD_ISTFT];             // 13 KB
  const int nt = blockIdx.y;
  {
    const uint4* src = (const uint4*)(ibt + (size_t)nt * 16 * KPAD_ISTFT);
    uint4* dst = (uint4*)bsh;
    for (int i = threadIdx.x; i < (16 * KPAD_ISTFT) / 8; i += 256) dst[i] = src[i];
  }
  int wave = threadIdx.x >> 5;
  int lane = threadIdx.x & 31;
  int hid = lane >> 4, lr = lane & 15;
  int mt0 = (blockIdx.x * 8 + wave) * 4;
  const f16_t* arow = ycat + (size_t)(mt0 * 16 + lr) * KPAD_ISTFT + hid * 8;
  __builtin_prefetch(arow, 0, 3);
  __builtin_prefetch(arow + 16 * KPAD_ISTFT, 0, 3);
  __builtin_prefetch(arow + 32 * KPAD_ISTFT, 0, 3);
  __builtin_prefetch(arow + 48 * KPAD_ISTFT, 0, 3);
  __syncthreads();

  const f16_t* bsrc = bsh + (size_t)lr * KPAD_ISTFT + hid * 8;
  v8f acc0 = {}, acc1 = {}, acc2 = {}, acc3 = {};
#pragma unroll
  for (int kt = 0; kt < KPAD_ISTFT / 32; ++kt) {
    v16h b = cat8(*(const v8h*)(bsrc + kt * 32), *(const v8h*)(bsrc + kt * 32 + 16));
    const f16_t* ak = arow + kt * 32;
    v16h a0 = cat8(*(const v8h*)(ak),                    *(const v8h*)(ak + 16));
    v16h a1 = cat8(*(const v8h*)(ak + 16 * KPAD_ISTFT),  *(const v8h*)(ak + 16 * KPAD_ISTFT + 16));
    v16h a2 = cat8(*(const v8h*)(ak + 32 * KPAD_ISTFT),  *(const v8h*)(ak + 32 * KPAD_ISTFT + 16));
    v16h a3 = cat8(*(const v8h*)(ak + 48 * KPAD_ISTFT),  *(const v8h*)(ak + 48 * KPAD_ISTFT + 16));
    acc0 = __builtin_amdgcn_wmma_f32_16x16x32_f16(false, a0, false, b, (short)0, acc0, false, false);
    acc1 = __builtin_amdgcn_wmma_f32_16x16x32_f16(false, a1, false, b, (short)0, acc1, false, false);
    acc2 = __builtin_amdgcn_wmma_f32_16x16x32_f16(false, a2, false, b, (short)0, acc2, false, false);
    acc3 = __builtin_amdgcn_wmma_f32_16x16x32_f16(false, a3, false, b, (short)0, acc3, false, false);
  }
  int row0 = hid * 8;
  int col = nt * 16 + lr;
  v8f accs[4] = {acc0, acc1, acc2, acc3};
#pragma unroll
  for (int u = 0; u < 4; ++u) {
    int mbase = (mt0 + u) * 16 + row0;
#pragma unroll
    for (int v = 0; v < 8; ++v)
      fout[(size_t)(mbase + v) * N_FFTC + col] = (f16_t)accs[u][v];
  }
}

// ---------------------------------------------------------------------------
// Stage 6: overlap-add + Hann^2 envelope divide + center crop.
// Every output sample has exactly two contributing frames (50% overlap).
// ---------------------------------------------------------------------------
__global__ void k_ola(const f16_t* __restrict__ fout, float* __restrict__ out) {
  int idx = blockIdx.x * blockDim.x + threadIdx.x;
  if (idx >= 2 * BATCH * OUTLEN) return;
  int p = idx % OUTLEN, bs = idx / OUTLEN;
  int g = p + HOPC;
  int f0 = g / HOPC;                           // in [1, FR-1]
  int n0 = g % HOPC;
  const f16_t* base = fout + (size_t)bs * FRPC * N_FFTC;
  float v = (float)base[(size_t)f0 * N_FFTC + n0] +
            (float)base[(size_t)(f0 - 1) * N_FFTC + n0 + HOPC];
  float w0 = 0.5f * (1.0f - cosf(TWO_PI * (float)n0 / (float)N_FFTC));
  float w1 = 0.5f * (1.0f - cosf(TWO_PI * (float)(n0 + HOPC) / (float)N_FFTC));
  out[idx] = v / (w0 * w0 + w1 * w1);          // envelope >= 0.5, no clamp needed
}

// ---------------------------------------------------------------------------
extern "C" void kernel_launch(void* const* d_in, const int* in_sizes, int n_in,
                              void* d_out, int out_size, void* d_ws, size_t ws_size,
                              hipStream_t stream) {
  const float* x    = (const float*)d_in[0];
  const float* lmr  = (const float*)d_in[1];
  const float* lmdr = (const float*)d_in[2];
  const float* ang  = (const float*)d_in[3];
  float* out = (float*)d_out;

  uint8_t* w = (uint8_t*)d_ws;
  size_t off = 0;
  auto carve = [&](size_t bytes) -> void* {
    off = (off + 255) & ~(size_t)255;
    void* p = w + off;
    off += bytes;
    return p;
  };
  f16_t* bst  = (f16_t*)carve((size_t)NPAD_STFT * N_FFTC * sizeof(f16_t));
  f16_t* ibt  = (f16_t*)carve((size_t)N_FFTC * KPAD_ISTFT * sizeof(f16_t));
  float* wr   = (float*)carve((size_t)CHC * STEPSC * sizeof(float));
  float* wi   = (float*)carve((size_t)CHC * STEPSC * sizeof(float));
  f16_t* fa   = (f16_t*)carve((size_t)MROWS * N_FFTC * sizeof(f16_t)); // reused as fout
  float* Hr   = (float*)carve((size_t)CHC * BATCH * FRPC * sizeof(float));
  float* Hi   = (float*)carve((size_t)CHC * BATCH * FRPC * sizeof(float));
  size_t ycat_bytes = (size_t)MROWS * KPAD_ISTFT * sizeof(f16_t);
  f16_t* ycat = (f16_t*)carve(ycat_bytes);

  k_weights<<<(FREQC * STEPSC + 255) / 256, 256, 0, stream>>>(lmr, lmdr, ang, wr, wi);
  k_stft_basis<<<(NPAD_STFT * N_FFTC + 255) / 256, 256, 0, stream>>>(bst);
  k_istft_basis<<<(N_FFTC * KPAD_ISTFT + 255) / 256, 256, 0, stream>>>(ibt);
  k_frames<<<(MROWS * N_FFTC + 255) / 256, 256, 0, stream>>>(x, fa);
  hipMemsetAsync(ycat, 0, ycat_bytes, stream);

  // STFT GEMM: 1376 M-tiles / (8 waves * 4 tiles) = 43 blocks; 25 N-tiles
  {
    dim3 grid(MTILES / 32, NPAD_STFT / 16);
    k_stft_wmma<<<grid, 256, 0, stream>>>(fa, bst, Hr, Hi);
  }

  // depthwise complex conv, one block per (channel, batch)
  k_conv<<<CHC * BATCH, 256, 0, stream>>>(Hr, Hi, wr, wi, ycat);

  // ISTFT GEMM: 43 blocks x 24 N-tiles (fout aliases fa; fa is dead by now)
  {
    dim3 grid(MTILES / 32, N_FFTC / 16);
    k_istft_wmma<<<grid, 256, 0, stream>>>(ycat, ibt, fa);
  }

  k_ola<<<(2 * BATCH * OUTLEN + 255) / 256, 256, 0, stream>>>(fa, out);
}
